// NLBlockND_87754771791958
// MI455X (gfx1250) — matrix operand order repaired and checked by reading, not compiled, not run on hardware
//
#include <hip/hip_runtime.h>
#include <hip/hip_bf16.h>

// ---------------------------------------------------------------------------
// NLBlockND on MI455X (gfx1250, wave32, WMMA).
// - All GEMMs: v_wmma_f32_16x16x32_bf16 (bf16 operands, f32 accumulate).
// - Flash-attention style softmax (f [B,4096,1024] never materialized).
// - BN stats via 128x128 Gram matrix of y (wy = Wz*y + b is linear).
// - All tiles staged K-contiguous so fragments load as 2x b128 + shuffle.
// - bf16 sources staged into LDS with async DMA (global_load_async_to_lds_b128).
// ---------------------------------------------------------------------------

typedef __attribute__((ext_vector_type(16))) __bf16 v16bf;
typedef __attribute__((ext_vector_type(8)))  __bf16 v8bf;
typedef __attribute__((ext_vector_type(2)))  __bf16 v2bf;
typedef __attribute__((ext_vector_type(8)))  float  v8f;

#define WMMA_BF16(A, B, C) \
    __builtin_amdgcn_wmma_f32_16x16x32_bf16(false, (A), false, (B), (short)0, (C), false, false)

#define SHUF16(lo, hi) \
    __builtin_shufflevector((lo), (hi), 0, 1, 2, 3, 4, 5, 6, 7, 8, 9, 10, 11, 12, 13, 14, 15)

constexpr int kB_ = 8;       // batch
constexpr int kC  = 512;     // channels
constexpr int kN  = 4096;    // H*W
constexpr int kM  = 1024;    // pooled spatial
constexpr int kIC = 64;      // inter channels (theta/phi)
constexpr int kGC = 128;     // g channels
constexpr int PK  = 40;      // padded row length for K=32 tiles (80B, 16B aligned)
constexpr int PK64 = 72;     // padded row length for K=64 tiles (144B, 16B aligned)

// CDNA5 async DMA: copy 16B global -> LDS, tracked by ASYNCcnt.
__device__ __forceinline__ void async_b128(const void* gsrc, void* ldst) {
    asm volatile("global_load_async_to_lds_b128 %0, %1, off"
                 :: "v"((unsigned)(unsigned long long)ldst),
                    "v"((unsigned long long)gsrc)
                 : "memory");
}
__device__ __forceinline__ void wait_async0() {
    asm volatile("s_wait_asynccnt 0x0" ::: "memory");
}

// A-frag (16x32 MxK): lane = row; two contiguous 8-elem K runs at klo, 16+klo.
__device__ __forceinline__ v16bf load_afrag(const __bf16* row, int lane) {
    const int klo = (lane & 16) ? 8 : 0;
    v8bf lo = *(const v8bf*)(row + klo);
    v8bf hi = *(const v8bf*)(row + 16 + klo);
    return SHUF16(lo, hi);
}
// B-frag (32x16 KxN): lane = col; two contiguous 8-elem K runs at klo, 16+klo.
__device__ __forceinline__ v16bf load_bfrag(const __bf16* row, int lane) {
    const int klo = (lane & 16) ? 8 : 0;
    v8bf lo = *(const v8bf*)(row + klo);
    v8bf hi = *(const v8bf*)(row + 16 + klo);
    return SHUF16(lo, hi);
}

// ---------------------------------------------------------------------------
__global__ void k_zero(float* __restrict__ p, int n) {
    int i = blockIdx.x * blockDim.x + threadIdx.x;
    if (i < n) p[i] = 0.0f;
}

// ---------------------------------------------------------------------------
// 1x1 conv as GEMM: out = W[O,C] * x[C,N] + bias. fp32 in, bf16 out.
// o_major=0: out[b][n][o]   (theta, phi: feeds A-frags / pooling over ic rows)
// o_major=1: out[b][o][n]   (g: feeds channel-major pooled layout)
// grid (N/128, O/16, B), block 256 = 8 waves, one 16x16 tile per wave.
// ---------------------------------------------------------------------------
__global__ __launch_bounds__(256) void k_proj(const float* __restrict__ x,
                                              const float* __restrict__ W,
                                              const float* __restrict__ bias,
                                              __bf16* __restrict__ out,
                                              int O, int o_major) {
    __shared__ __align__(16) __bf16 sXT[128][PK];  // [n][K]  (B operand)
    __shared__ __align__(16) __bf16 sW[16][PK];    // [o][K]  (A operand)
    const int tid = threadIdx.x, lane = tid & 31, wv = tid >> 5;
    const int nb = blockIdx.x * 128, ob = blockIdx.y * 16, b = blockIdx.z;
    const float* xb = x + (size_t)b * kC * kN;
    const int col = lane & 15;
    v8f acc = {};
    for (int k0 = 0; k0 < kC; k0 += 32) {
        // stage x chunk transposed [n][K], packing pairs into b32 stores
        for (int i = tid; i < 16 * 128; i += 256) {
            int kp = i >> 7, nn = i & 127;
            int kk = kp * 2;
            v2bf t;
            t[0] = (__bf16)xb[(size_t)(k0 + kk) * kN + nb + nn];
            t[1] = (__bf16)xb[(size_t)(k0 + kk + 1) * kN + nb + nn];
            *(v2bf*)&sXT[nn][kk] = t;
        }
        for (int i = tid; i < 16 * 32; i += 256) {
            int oo = i >> 5, kk = i & 31;
            sW[oo][kk] = (__bf16)W[(size_t)(ob + oo) * kC + k0 + kk];
        }
        __syncthreads();
        v16bf a  = load_afrag(&sW[col][0], lane);
        v16bf bm = load_bfrag(&sXT[wv * 16 + col][0], lane);
        acc = WMMA_BF16(a, bm, acc);
        __syncthreads();
    }
    const int hi = (lane >> 4) & 1;
    const int n = nb + wv * 16 + col;
    if (!o_major) {
        v8bf ov;
#pragma unroll
        for (int k = 0; k < 8; k++) ov[k] = (__bf16)(acc[k] + bias[ob + k + 8 * hi]);
        *(v8bf*)(out + (size_t)b * kN * O + (size_t)n * O + ob + 8 * hi) = ov;
    } else {
#pragma unroll
        for (int k = 0; k < 8; k++) {
            int o = ob + k + 8 * hi;
            out[((size_t)b * O + o) * kN + n] = (__bf16)(acc[k] + bias[o]);
        }
    }
}

// ---------------------------------------------------------------------------
// 2x2 maxpool, channel-last: [b][n][ch] -> [b][m][ch]   (phi)
// ---------------------------------------------------------------------------
__global__ void k_pool(const __bf16* __restrict__ in, __bf16* __restrict__ out,
                       int ch, int total) {
    int idx = blockIdx.x * blockDim.x + threadIdx.x;
    if (idx >= total) return;
    int c = idx % ch, m = (idx / ch) & 1023, b = idx / (ch * 1024);
    int hp = m >> 5, wp = m & 31;
    int n0 = (hp * 2) * 64 + wp * 2;
    const __bf16* ib = in + (size_t)b * kN * ch;
    float v0 = (float)ib[(size_t)n0 * ch + c];
    float v1 = (float)ib[(size_t)(n0 + 1) * ch + c];
    float v2 = (float)ib[(size_t)(n0 + 64) * ch + c];
    float v3 = (float)ib[(size_t)(n0 + 65) * ch + c];
    out[(size_t)b * 1024 * ch + (size_t)m * ch + c] =
        (__bf16)fmaxf(fmaxf(v0, v1), fmaxf(v2, v3));
}

// 2x2 maxpool, channel-major: [b][ch][n] -> [b][ch][m]   (g)
__global__ void k_poolT(const __bf16* __restrict__ in, __bf16* __restrict__ out,
                        int ch, int total) {
    int idx = blockIdx.x * blockDim.x + threadIdx.x;
    if (idx >= total) return;
    int m = idx & 1023, c = (idx >> 10) % ch, b = idx / (ch * 1024);
    int hp = m >> 5, wp = m & 31;
    int n0 = (hp * 2) * 64 + wp * 2;
    const __bf16* ib = in + ((size_t)b * ch + c) * kN;
    float v0 = (float)ib[n0], v1 = (float)ib[n0 + 1];
    float v2 = (float)ib[n0 + 64], v3 = (float)ib[n0 + 65];
    out[((size_t)b * ch + c) * 1024 + m] = (__bf16)fmaxf(fmaxf(v0, v1), fmaxf(v2, v3));
}

// ---------------------------------------------------------------------------
// Flash attention: y = softmax_m(theta·phi^T) · g
// theta [b][n][ic], phi [b][m][ic], g [b][gc][m]
// y written twice: y1 [b][gc][n] (for Gram) and y2 [b][n][gc] (for Wz GEMM).
// grid (N/128, B), block 256; each wave owns a 16-row n-block.
// ---------------------------------------------------------------------------
__global__ __launch_bounds__(256) void k_attn(const __bf16* __restrict__ theta,
                                              const __bf16* __restrict__ phi,
                                              const __bf16* __restrict__ g,
                                              __bf16* __restrict__ y1,
                                              __bf16* __restrict__ y2) {
    __shared__ __align__(16) __bf16 sPhi[32][PK64];   // [m][ic]      (B of f)
    __shared__ __align__(16) __bf16 sGT[128][PK];     // [gc][m]      (B of y)
    __shared__ __align__(16) __bf16 sP[8][16][PK];    // per-wave P [n][m] (A of y)
    const int tid = threadIdx.x, lane = tid & 31, wv = tid >> 5;
    const int b = blockIdx.y;
    const int nb = blockIdx.x * 128 + wv * 16;
    const int col = lane & 15, hi = (lane >> 4) & 1;
    const int klo = hi ? 8 : 0;

    // theta A-frags (K=64 -> two 16x32 frags) straight from global, 4x b128
    const __bf16* th = theta + (size_t)b * kN * kIC + (size_t)(nb + col) * kIC;
    v16bf a0 = SHUF16(*(const v8bf*)(th + klo), *(const v8bf*)(th + 16 + klo));
    v16bf a1 = SHUF16(*(const v8bf*)(th + 32 + klo), *(const v8bf*)(th + 48 + klo));

    v8f yacc[8];
    const v8f vzero = {};
#pragma unroll
    for (int t = 0; t < 8; t++) yacc[t] = vzero;
    float mi[8], li[8];
#pragma unroll
    for (int k = 0; k < 8; k++) { mi[k] = -3.0e38f; li[k] = 0.0f; }
    const float L2E = 1.44269504088896340736f;

    const __bf16* phiB = phi + (size_t)b * kM * kIC;
    const __bf16* gB   = g + (size_t)b * kGC * kM;

    for (int mc = 0; mc < kM; mc += 32) {
        // async DMA staging: phi rows (32 x 128B) and g rows (128 x 64B)
        {
            int m = tid >> 3, seg = tid & 7;  // 256 transfers
            async_b128(phiB + (size_t)(mc + m) * kIC + seg * 8, &sPhi[m][seg * 8]);
        }
#pragma unroll
        for (int j = 0; j < 2; j++) {         // 512 transfers
            int t = tid + 256 * j;
            int gc = t >> 2, seg = t & 3;
            async_b128(gB + (size_t)gc * kM + mc + seg * 8, &sGT[gc][seg * 8]);
        }
        wait_async0();
        __syncthreads();

        // f tiles (rows n, cols m): two 16-col subtiles, K(ic)=64
        v16bf b0lo = load_bfrag(&sPhi[col][0], lane);
        v16bf b0hi = load_bfrag(&sPhi[col][32], lane);
        v16bf b1lo = load_bfrag(&sPhi[col + 16][0], lane);
        v16bf b1hi = load_bfrag(&sPhi[col + 16][32], lane);
        v8f f0 = {}, f1 = {};
        f0 = WMMA_BF16(a0, b0lo, f0);
        f0 = WMMA_BF16(a1, b0hi, f0);
        f1 = WMMA_BF16(a0, b1lo, f1);
        f1 = WMMA_BF16(a1, b1hi, f1);

        // online softmax (row = k + 8*hi lives within a 16-lane half)
        float p0[8], p1[8];
#pragma unroll
        for (int k = 0; k < 8; k++) {
            float tm = fmaxf(f0[k], f1[k]);
#pragma unroll
            for (int msk = 1; msk < 16; msk <<= 1)
                tm = fmaxf(tm, __shfl_xor(tm, msk, 32));
            float mn = fmaxf(mi[k], tm);
            float corr = __builtin_amdgcn_exp2f((mi[k] - mn) * L2E);
            p0[k] = __builtin_amdgcn_exp2f((f0[k] - mn) * L2E);
            p1[k] = __builtin_amdgcn_exp2f((f1[k] - mn) * L2E);
            float rs = p0[k] + p1[k];
#pragma unroll
            for (int msk = 1; msk < 16; msk <<= 1)
                rs += __shfl_xor(rs, msk, 32);
            li[k] = li[k] * corr + rs;
            mi[k] = mn;
#pragma unroll
            for (int t = 0; t < 8; t++) yacc[t][k] *= corr;
        }

        // C-layout -> A-layout transpose of P through wave-private LDS
#pragma unroll
        for (int k = 0; k < 8; k++) {
            sP[wv][k + 8 * hi][col]      = (__bf16)p0[k];
            sP[wv][k + 8 * hi][col + 16] = (__bf16)p1[k];
        }
        v16bf pa = load_afrag(&sP[wv][col][0], lane);

#pragma unroll
        for (int t = 0; t < 8; t++) {
            v16bf gb = load_bfrag(&sGT[16 * t + col][0], lane);
            yacc[t] = WMMA_BF16(pa, gb, yacc[t]);
        }
        __syncthreads();
    }

    // finalize: divide by row sums; dual-layout stores
    float inv[8];
#pragma unroll
    for (int k = 0; k < 8; k++) inv[k] = 1.0f / li[k];
#pragma unroll
    for (int t = 0; t < 8; t++) {
        int gc = 16 * t + col;
        v8bf yv;
#pragma unroll
        for (int k = 0; k < 8; k++) yv[k] = (__bf16)(yacc[t][k] * inv[k]);
        *(v8bf*)(y1 + ((size_t)b * kGC + gc) * kN + nb + 8 * hi) = yv;
    }
#pragma unroll
    for (int k = 0; k < 8; k++) {
        __bf16* p = y2 + ((size_t)b * kN + nb + k + 8 * hi) * kGC + col;
#pragma unroll
        for (int t = 0; t < 8; t++) p[16 * t] = (__bf16)(yacc[t][k] * inv[k]);
    }
}

// ---------------------------------------------------------------------------
// Gram matrix G = Y Y^T (128x128) + ybar, split-K with f32 atomics.
// y1 [b][gc][n]; grid (N/512, B), block 256; wave w owns row-tile w.
// ---------------------------------------------------------------------------
__global__ __launch_bounds__(256) void k_gram(const __bf16* __restrict__ y1,
                                              float* __restrict__ G,
                                              float* __restrict__ ybar) {
    __shared__ __align__(16) __bf16 sY[128][PK];  // [gc][n-chunk]
    const int tid = threadIdx.x, lane = tid & 31, wv = tid >> 5;
    const int b = blockIdx.y, nc0 = blockIdx.x * 512;
    const int col = lane & 15, hi = (lane >> 4) & 1;
    const __bf16* yB = y1 + (size_t)b * kGC * kN;
    v8f acc[8];
    const v8f vzero = {};
#pragma unroll
    for (int j = 0; j < 8; j++) acc[j] = vzero;
    float ysum = 0.0f;
    for (int ks = 0; ks < 512; ks += 32) {
#pragma unroll
        for (int j = 0; j < 2; j++) {  // 512 x 16B transfers
            int t = tid + 256 * j;
            int gc = t >> 2, seg = t & 3;
            async_b128(yB + (size_t)gc * kN + nc0 + ks + seg * 8, &sY[gc][seg * 8]);
        }
        wait_async0();
        __syncthreads();
        if (tid < 128) {
            float s = 0.0f;
#pragma unroll
            for (int v = 0; v < 4; v++) {
                v8bf r = *(const v8bf*)&sY[tid][v * 8];
#pragma unroll
                for (int e = 0; e < 8; e++) s += (float)r[e];
            }
            ysum += s;
        }
        v16bf ai = load_afrag(&sY[16 * wv + col][0], lane);
#pragma unroll
        for (int j = 0; j < 8; j++) {
            v16bf bj = load_bfrag(&sY[16 * j + col][0], lane);
            acc[j] = WMMA_BF16(ai, bj, acc[j]);
        }
        __syncthreads();
    }
#pragma unroll
    for (int j = 0; j < 8; j++)
#pragma unroll
        for (int k = 0; k < 8; k++)
            atomicAdd(&G[(16 * wv + k + 8 * hi) * kGC + 16 * j + col], acc[j][k]);
    if (tid < 128) atomicAdd(&ybar[tid], ysum);
}

// ---------------------------------------------------------------------------
// Per-channel BN scale/shift from Gram + ybar.
// ---------------------------------------------------------------------------
__global__ void k_bnstats(const float* __restrict__ wz_w, const float* __restrict__ wz_b,
                          const float* __restrict__ gamma, const float* __restrict__ beta,
                          const float* __restrict__ G, const float* __restrict__ ybar,
                          float* __restrict__ scale, float* __restrict__ shift) {
    int o = blockIdx.x * blockDim.x + threadIdx.x;
    if (o >= kC) return;
    const float* w = wz_w + (size_t)o * kGC;
    const float inv = 1.0f / ((float)kB_ * (float)kN);
    float s1 = 0.0f;
    for (int i = 0; i < kGC; i++) s1 += w[i] * ybar[i];
    float q = 0.0f;
    for (int i = 0; i < kGC; i++) {
        float t = 0.0f;
        for (int j = 0; j < kGC; j++) t += G[i * kGC + j] * w[j];
        q += w[i] * t;
    }
    float bo = wz_b[o];
    float mean = s1 * inv + bo;
    float e2 = q * inv + 2.0f * bo * (s1 * inv) + bo * bo;
    float var = e2 - mean * mean;
    float sc = gamma[o] * rsqrtf(var + 1e-5f);
    scale[o] = sc;
    shift[o] = beta[o] - mean * sc;
}

// ---------------------------------------------------------------------------
// Wz GEMM (K=128) fused with BN affine + residual. y2 [b][n][gc], fp32 out.
// ---------------------------------------------------------------------------
__global__ __launch_bounds__(256) void k_out(const float* __restrict__ wz_w,
                                             const __bf16* __restrict__ y2,
                                             const float* __restrict__ wz_b,
                                             const float* __restrict__ scale,
                                             const float* __restrict__ shift,
                                             const float* __restrict__ x,
                                             float* __restrict__ out) {
    __shared__ __align__(16) __bf16 sBN[128][PK];  // [n][gc-chunk]  (B operand)
    __shared__ __align__(16) __bf16 sA[16][PK];    // [o][gc-chunk]  (A operand)
    const int tid = threadIdx.x, lane = tid & 31, wv = tid >> 5;
    const int nb = blockIdx.x * 128, ob = blockIdx.y * 16, b = blockIdx.z;
    const int col = lane & 15;
    const __bf16* yB = y2 + (size_t)b * kN * kGC;
    v8f acc = {};
    for (int k0 = 0; k0 < kGC; k0 += 32) {
#pragma unroll
        for (int j = 0; j < 2; j++) {  // async DMA: 128 rows x 64B
            int t = tid + 256 * j;
            int nn = t >> 2, seg = t & 3;
            async_b128(yB + (size_t)(nb + nn) * kGC + k0 + seg * 8, &sBN[nn][seg * 8]);
        }
        for (int i = tid; i < 16 * 32; i += 256) {
            int oo = i >> 5, kk = i & 31;
            sA[oo][kk] = (__bf16)wz_w[(size_t)(ob + oo) * kGC + k0 + kk];
        }
        wait_async0();
        __syncthreads();
        v16bf a  = load_afrag(&sA[col][0], lane);
        v16bf bm = load_bfrag(&sBN[wv * 16 + col][0], lane);
        acc = WMMA_BF16(a, bm, acc);
        __syncthreads();
    }
    const int hi = (lane >> 4) & 1;
    const int n = nb + wv * 16 + col;
#pragma unroll
    for (int k = 0; k < 8; k++) {
        int o = ob + k + 8 * hi;
        float wy = acc[k] + wz_b[o];
        size_t idx = ((size_t)b * kC + o) * kN + n;
        out[idx] = scale[o] * wy + shift[o] + x[idx];
    }
}

// ---------------------------------------------------------------------------
extern "C" void kernel_launch(void* const* d_in, const int* in_sizes, int n_in,
                              void* d_out, int out_size, void* d_ws, size_t ws_size,
                              hipStream_t stream) {
    const float* x       = (const float*)d_in[0];
    const float* g_w     = (const float*)d_in[1];
    const float* g_b     = (const float*)d_in[2];
    const float* theta_w = (const float*)d_in[3];
    const float* theta_b = (const float*)d_in[4];
    const float* phi_w   = (const float*)d_in[5];
    const float* phi_b   = (const float*)d_in[6];
    const float* wz_w    = (const float*)d_in[7];
    const float* wz_b    = (const float*)d_in[8];
    const float* gamma   = (const float*)d_in[9];
    const float* beta    = (const float*)d_in[10];
    float* out = (float*)d_out;

    char* ws = (char*)d_ws;
    size_t off = 0;
    auto alloc = [&](size_t bytes) { void* p = ws + off; off = (off + bytes + 255) & ~(size_t)255; return p; };
    __bf16* thetaB = (__bf16*)alloc((size_t)kB_ * kN * kIC * 2);  // [b][n][ic]
    __bf16* phiF   = (__bf16*)alloc((size_t)kB_ * kN * kIC * 2);  // [b][n][ic]
    __bf16* gF     = (__bf16*)alloc((size_t)kB_ * kGC * kN * 2);  // [b][gc][n]
    __bf16* phiP   = (__bf16*)alloc((size_t)kB_ * kM * kIC * 2);  // [b][m][ic]
    __bf16* gP     = (__bf16*)alloc((size_t)kB_ * kGC * kM * 2);  // [b][gc][m]
    __bf16* y1     = (__bf16*)alloc((size_t)kB_ * kGC * kN * 2);  // [b][gc][n]
    __bf16* y2     = (__bf16*)alloc((size_t)kB_ * kN * kGC * 2);  // [b][n][gc]
    float*  G      = (float*)alloc((size_t)kGC * kGC * 4);
    float*  ybar   = (float*)alloc((size_t)kGC * 4);
    float*  scale  = (float*)alloc((size_t)kC * 4);
    float*  shift  = (float*)alloc((size_t)kC * 4);

    int nz = kGC * kGC + kGC;  // G + ybar contiguous
    k_zero<<<(nz + 255) / 256, 256, 0, stream>>>(G, nz);

    k_proj<<<dim3(kN / 128, kIC / 16, kB_), 256, 0, stream>>>(x, theta_w, theta_b, thetaB, kIC, 0);
    k_proj<<<dim3(kN / 128, kIC / 16, kB_), 256, 0, stream>>>(x, phi_w, phi_b, phiF, kIC, 0);
    k_proj<<<dim3(kN / 128, kGC / 16, kB_), 256, 0, stream>>>(x, g_w, g_b, gF, kGC, 1);

    int tphi = kB_ * kM * kIC;
    k_pool<<<(tphi + 255) / 256, 256, 0, stream>>>(phiF, phiP, kIC, tphi);
    int tg = kB_ * kGC * kM;
    k_poolT<<<(tg + 255) / 256, 256, 0, stream>>>(gF, gP, kGC, tg);

    k_attn<<<dim3(kN / 128, kB_), 256, 0, stream>>>(thetaB, phiP, gP, y1, y2);

    k_gram<<<dim3(kN / 512, kB_), 256, 0, stream>>>(y1, G, ybar);
    k_bnstats<<<2, 256, 0, stream>>>(wz_w, wz_b, gamma, beta, G, ybar, scale, shift);

    k_out<<<dim3(kN / 128, kC / 16, kB_), 256, 0, stream>>>(wz_w, y2, wz_b, scale, shift, x, out);
}
